// BayesianDropoutLSTM_52158082842916
// MI455X (gfx1250) — compile-verified, matrix operand order checked by reading, not run on hardware
//
#include <hip/hip_runtime.h>
#include <hip/hip_bf16.h>

// ---------------- problem constants ----------------
constexpr int VOCAB = 50000;
constexpr int TAG   = 48;
constexpr int T_    = 237;
constexpr int D_    = 512;
constexpr int H_    = 512;
constexpr int B_    = 512;
constexpr int KW_   = H_ + D_;        // 1024 : concat [W_hh | W_ih] K-extent

typedef __attribute__((ext_vector_type(16))) __bf16 v16bf;
typedef __attribute__((ext_vector_type(8)))  float  v8f;

union TileBits { uint4 q[2]; v16bf v; };

// round-to-nearest-even f32 -> bf16 (bit pattern)
__device__ __forceinline__ unsigned short f32_to_bf16(float f) {
  union { float f; unsigned int u; } v; v.f = f;
  unsigned int lsb = (v.u >> 16) & 1u;
  unsigned int r = v.u + 0x7fffu + lsb;
  return (unsigned short)(r >> 16);
}

__device__ __forceinline__ float sigmoidf_(float x) {
  return 1.0f / (1.0f + __expf(-x));
}

// Load one 16x32 bf16 tile (row-major source) into WMMA A/B per-lane layout:
// lane L holds row (L&15); halves 0..7 = K [ko..ko+7], halves 8..15 = K [ko+16..ko+23]
// with ko = (L>=16) ? 8 : 0.  base points at element [row0][k0]; stride in elements.
__device__ __forceinline__ v16bf load_tile16x32(const unsigned short* __restrict__ base,
                                                int stride, int lane) {
  const int r  = lane & 15;
  const int ko = (lane >> 4) ? 8 : 0;
  const unsigned short* p = base + (size_t)r * stride + ko;
  TileBits t;
  t.q[0] = *reinterpret_cast<const uint4*>(p);
  t.q[1] = *reinterpret_cast<const uint4*>(p + 16);
  return t.v;
}

__device__ __forceinline__ v8f zero_v8f() {
  v8f z;
#pragma unroll
  for (int i = 0; i < 8; ++i) z[i] = 0.0f;
  return z;
}

// ---------------- pre-processing kernels ----------------

__global__ void cvt_f32_to_bf16_kernel(unsigned short* __restrict__ dst,
                                       const float* __restrict__ src, int n) {
  int i = blockIdx.x * blockDim.x + threadIdx.x;
  int step = gridDim.x * blockDim.x;
  for (; i < n; i += step) dst[i] = f32_to_bf16(src[i]);
}

// Wcat[g][k] : k<H -> W_hh[g][k] ; k>=H -> W_ih[g][k-H]   (bf16, row-major, ld=KW_)
__global__ void build_wcat_kernel(unsigned short* __restrict__ dst,
                                  const float* __restrict__ W_hh,
                                  const float* __restrict__ W_ih) {
  int i = blockIdx.x * blockDim.x + threadIdx.x;
  int step = gridDim.x * blockDim.x;
  const int total = 4 * H_ * KW_;
  for (; i < total; i += step) {
    int g = i / KW_;
    int k = i - g * KW_;
    float v = (k < H_) ? W_hh[(size_t)g * H_ + k] : W_ih[(size_t)g * D_ + (k - H_)];
    dst[i] = f32_to_bf16(v);
  }
}

__global__ void init_state_kernel(unsigned int* __restrict__ h0,  // bf16 pairs
                                  float* __restrict__ c, int nh, int nc) {
  int i = blockIdx.x * blockDim.x + threadIdx.x;
  int step = gridDim.x * blockDim.x;
  for (int j = i; j < nh; j += step) h0[j] = 0u;
  for (int j = i; j < nc; j += step) c[j] = 0.0f;
}

// ---------------- fused LSTM step ----------------
// One wave per block. Block computes h_new/c_new for 16 batch rows x 32 h-cols.
// gates[b, g] = sum_k [h | emb] [b,k] * Wcat[g,k]  (bf16 WMMA, f32 acc)
__global__ __launch_bounds__(32)
void lstm_step_kernel(const unsigned short* __restrict__ h_in,   // [B][H] bf16
                      unsigned short* __restrict__ h_out,        // [B][H] bf16
                      float* __restrict__ c,                     // [B][H] f32 (in place)
                      const unsigned short* __restrict__ Ebf,    // [VOCAB][D] bf16
                      const unsigned short* __restrict__ Wcat,   // [4H][KW] bf16
                      const int* __restrict__ x,                 // [B][T]
                      const int* __restrict__ lens,              // [B]
                      const float* __restrict__ b_ih,
                      const float* __restrict__ b_hh,
                      unsigned short* __restrict__ outs,         // [B*T][H] bf16
                      int t) {
  const int lane = threadIdx.x;
  const int b0   = blockIdx.y * 16;   // batch tile
  const int hc0  = blockIdx.x * 32;   // hidden-col tile

  v8f acc[4][2];
#pragma unroll
  for (int gi = 0; gi < 4; ++gi)
#pragma unroll
    for (int s = 0; s < 2; ++s) acc[gi][s] = zero_v8f();

  // ---- K part 1: recurrent h contribution (K = 0..H) ----
  for (int k0 = 0; k0 < H_; k0 += 32) {
    v16bf a = load_tile16x32(h_in + (size_t)b0 * H_ + k0, H_, lane);
#pragma unroll
    for (int gi = 0; gi < 4; ++gi) {
#pragma unroll
      for (int s = 0; s < 2; ++s) {
        const unsigned short* wb = Wcat + (size_t)(gi * H_ + hc0 + s * 16) * KW_ + k0;
        v16bf bt = load_tile16x32(wb, KW_, lane);
        acc[gi][s] = __builtin_amdgcn_wmma_f32_16x16x32_bf16(
            false, a, false, bt, (short)0, acc[gi][s], false, false);
      }
    }
  }

  // ---- K part 2: gathered embedding contribution (K = H..H+D) ----
  const int tok = x[(size_t)(b0 + (lane & 15)) * T_ + t];
  const unsigned short* erow = Ebf + (size_t)tok * D_ + ((lane >> 4) ? 8 : 0);
  for (int k0 = 0; k0 < D_; k0 += 32) {
    TileBits ta;
    ta.q[0] = *reinterpret_cast<const uint4*>(erow + k0);
    ta.q[1] = *reinterpret_cast<const uint4*>(erow + k0 + 16);
    v16bf a = ta.v;
#pragma unroll
    for (int gi = 0; gi < 4; ++gi) {
#pragma unroll
      for (int s = 0; s < 2; ++s) {
        const unsigned short* wb =
            Wcat + (size_t)(gi * H_ + hc0 + s * 16) * KW_ + (H_ + k0);
        v16bf bt = load_tile16x32(wb, KW_, lane);
        acc[gi][s] = __builtin_amdgcn_wmma_f32_16x16x32_bf16(
            false, a, false, bt, (short)0, acc[gi][s], false, false);
      }
    }
  }

  // ---- epilogue: gates -> (h, c, out) with pack_padded masking ----
  // C/D layout: element (row = v + 8*(lane>>4), col = lane&15) in acc[*][s][v]
  const int n     = lane & 15;
  const int rbase = (lane >> 4) * 8;
#pragma unroll
  for (int s = 0; s < 2; ++s) {
    const int hc = hc0 + s * 16 + n;
    const float bi = b_ih[0 * H_ + hc] + b_hh[0 * H_ + hc];
    const float bf = b_ih[1 * H_ + hc] + b_hh[1 * H_ + hc];
    const float bg = b_ih[2 * H_ + hc] + b_hh[2 * H_ + hc];
    const float bo = b_ih[3 * H_ + hc] + b_hh[3 * H_ + hc];
#pragma unroll
    for (int v = 0; v < 8; ++v) {
      const int b = b0 + rbase + v;
      const float iv = sigmoidf_(acc[0][s][v] + bi);
      const float fv = sigmoidf_(acc[1][s][v] + bf);
      const float gv = tanhf(acc[2][s][v] + bg);
      const float ov = sigmoidf_(acc[3][s][v] + bo);
      const size_t idx = (size_t)b * H_ + hc;
      const float cp = c[idx];
      const float cn = fv * cp + iv * gv;
      const float hn = ov * tanhf(cn);
      const bool  m  = t < lens[b];
      c[idx] = m ? cn : cp;
      const unsigned short hb = f32_to_bf16(hn);
      h_out[idx] = m ? hb : h_in[idx];
      outs[((size_t)b * T_ + t) * H_ + hc] = m ? hb : (unsigned short)0;
    }
  }
}

// ---------------- final FC: [B*T, H] @ fc_W^T + fc_b -> [B*T, 48] f32 ----------------
__global__ __launch_bounds__(32)
void fc_kernel(const unsigned short* __restrict__ outs,   // [R][H] bf16
               const unsigned short* __restrict__ fcWbf,  // [TAG][H] bf16
               const float* __restrict__ fc_b,
               float* __restrict__ out) {
  const int lane = threadIdx.x;
  const int r0   = blockIdx.x * 16;

  v8f acc[3];
#pragma unroll
  for (int s = 0; s < 3; ++s) acc[s] = zero_v8f();

  for (int k0 = 0; k0 < H_; k0 += 32) {
    v16bf a = load_tile16x32(outs + (size_t)r0 * H_ + k0, H_, lane);
#pragma unroll
    for (int s = 0; s < 3; ++s) {
      v16bf bt = load_tile16x32(fcWbf + (size_t)(s * 16) * H_ + k0, H_, lane);
      acc[s] = __builtin_amdgcn_wmma_f32_16x16x32_bf16(
          false, a, false, bt, (short)0, acc[s], false, false);
    }
  }

  const int n     = lane & 15;
  const int rbase = (lane >> 4) * 8;
#pragma unroll
  for (int s = 0; s < 3; ++s) {
    const int col = s * 16 + n;
    const float bias = fc_b[col];
#pragma unroll
    for (int v = 0; v < 8; ++v) {
      const int r = r0 + rbase + v;
      out[(size_t)r * TAG + col] = acc[s][v] + bias;
    }
  }
}

// ---------------- host launch ----------------
extern "C" void kernel_launch(void* const* d_in, const int* in_sizes, int n_in,
                              void* d_out, int out_size, void* d_ws, size_t ws_size,
                              hipStream_t stream) {
  const int*   x     = (const int*)d_in[0];
  const int*   lens  = (const int*)d_in[1];
  const float* E     = (const float*)d_in[2];
  const float* W_ih  = (const float*)d_in[3];
  const float* W_hh  = (const float*)d_in[4];
  const float* b_ih  = (const float*)d_in[5];
  const float* b_hh  = (const float*)d_in[6];
  const float* fc_W  = (const float*)d_in[7];
  const float* fc_b  = (const float*)d_in[8];
  float*       out   = (float*)d_out;

  // workspace layout (256B aligned)
  char* ws = (char*)d_ws;
  size_t off = 0;
  auto take = [&](size_t bytes) {
    size_t o = off;
    off = (off + bytes + 255) & ~(size_t)255;
    return o;
  };
  unsigned short* Ebf   = (unsigned short*)(ws + take((size_t)VOCAB * D_ * 2));
  unsigned short* Wcat  = (unsigned short*)(ws + take((size_t)4 * H_ * KW_ * 2));
  unsigned short* fcWbf = (unsigned short*)(ws + take((size_t)TAG * H_ * 2));
  unsigned short* hA    = (unsigned short*)(ws + take((size_t)B_ * H_ * 2));
  unsigned short* hB    = (unsigned short*)(ws + take((size_t)B_ * H_ * 2));
  float*          cbuf  = (float*)(ws + take((size_t)B_ * H_ * 4));
  unsigned short* outs  = (unsigned short*)(ws + take((size_t)B_ * T_ * H_ * 2));
  (void)ws_size; (void)in_sizes; (void)n_in; (void)out_size;

  // 1) precision conversion (one-time per call)
  {
    const int nE = VOCAB * D_;
    cvt_f32_to_bf16_kernel<<<4096, 256, 0, stream>>>(Ebf, E, nE);
    build_wcat_kernel<<<2048, 256, 0, stream>>>(Wcat, W_hh, W_ih);
    cvt_f32_to_bf16_kernel<<<96, 256, 0, stream>>>(fcWbf, fc_W, TAG * H_);
    init_state_kernel<<<1024, 256, 0, stream>>>((unsigned int*)hA, cbuf,
                                                B_ * H_ / 2, B_ * H_);
  }

  // 2) sequential LSTM: one fused kernel per timestep (ping-pong h)
  dim3 sgrid(H_ / 32, B_ / 16);  // 16 x 32 = 512 waves per step
  for (int t = 0; t < T_; ++t) {
    const unsigned short* hin = (t & 1) ? hB : hA;
    unsigned short*       hout = (t & 1) ? hA : hB;
    lstm_step_kernel<<<sgrid, 32, 0, stream>>>(hin, hout, cbuf, Ebf, Wcat,
                                               x, lens, b_ih, b_hh, outs, t);
  }

  // 3) final FC
  fc_kernel<<<(B_ * T_) / 16, 32, 0, stream>>>(outs, fcWbf, fc_b, out);
}
